// MaSIF_ligand_site_66314295050831
// MI455X (gfx1250) — matrix operand order repaired and checked by reading, not compiled, not run on hardware
//
#include <hip/hip_runtime.h>

#define TWO_PI_F 6.283185307179586f

typedef _Float16 v16h __attribute__((ext_vector_type(16)));
typedef _Float16 v8h  __attribute__((ext_vector_type(8)));
typedef float    v8f  __attribute__((ext_vector_type(8)));

constexpr int B_   = 4;
constexpr int S_   = 256;
constexpr int V_   = 200;
constexpr int F_   = 4;
constexpr int G_   = 80;
constexpr int BS   = B_ * S_;      // 1024
constexpr int NROT = 16;
constexpr int KPAD = 96;           // 80 padded to multiple of 32
constexpr int FG   = F_ * G_;      // 320

// ---------------------------------------------------------------------------
// A-fragment loader for v_wmma_f32_16x16x32_f16.
// A 16x32 k order used here (contiguous, B packed to match): lanes 0-15 row M=lane, K = 0..15;
// lanes 16-31 row M=lane-16, K = 16..31.  Two 16B loads per lane.
// ---------------------------------------------------------------------------
__device__ inline v16h load_a16x32(const _Float16* A, int lda, int row0, int k0, int lane) {
  int row = row0 + (lane & 15);
  int kh  = (lane & 16) ? 16 : 0;
  const _Float16* p = A + (size_t)row * lda + k0 + kh;
  v8h lo = *(const v8h*)p;          // K = kh .. kh+7
  v8h hi = *(const v8h*)(p + 8);    // K = kh+8 .. kh+15
  v16h a;
#pragma unroll
  for (int i = 0; i < 8; ++i) { a[i] = lo[i]; a[8 + i] = hi[i]; }
  return a;
}

// ---------------------------------------------------------------------------
// Prep: swizzle weights into B-fragment layout (lane = column n, 16 contiguous
// K values per lane: lanes 0-15 K=0..15, lanes 16-31 K=16..31), compute BN
// scale/shift, zero the x1 pad columns.  Runs once per launch.
// ---------------------------------------------------------------------------
__global__ __launch_bounds__(256) void prep_kernel(
    const float* __restrict__ Wconv, const float* __restrict__ d1w,
    const float* __restrict__ d2w,
    const float* __restrict__ gamma, const float* __restrict__ beta,
    const float* __restrict__ mean,  const float* __restrict__ var,
    _Float16* __restrict__ WcF, _Float16* __restrict__ D1F,
    _Float16* __restrict__ D2F,
    float* __restrict__ bnScale, float* __restrict__ bnShift,
    _Float16* __restrict__ x1)
{
  int id = blockIdx.x * 256 + threadIdx.x;
  const int N_WC = 60 * 512;   // F*3kt*5nt tiles
  const int N_D1 = 50 * 512;   // 10kt*5nt tiles
  const int N_D2 = 3 * 512;    // 3kt tiles
  const int N_BN = 2 * FG;
  const int N_XP = BS * 16;

  if (id < N_WC) {
    int e = id & 15, lane = (id >> 4) & 31, t = id >> 9;
    int n = t % 5, kt = (t / 5) % 3, f = t / 15;
    int k = kt * 32 + ((lane >= 16) ? 16 : 0) + e;
    int h = n * 16 + (lane & 15);
    float v = (k < G_) ? Wconv[(f * G_ + k) * G_ + h] : 0.f;
    WcF[id] = (_Float16)v;
    return;
  }
  id -= N_WC;
  if (id < N_D1) {
    int e = id & 15, lane = (id >> 4) & 31, t = id >> 9;
    int n = t % 5, kt = t / 5;
    int k = kt * 32 + ((lane >= 16) ? 16 : 0) + e;   // < 320 always
    int h = n * 16 + (lane & 15);
    D1F[id] = (_Float16)d1w[k * G_ + h];
    return;
  }
  id -= N_D1;
  if (id < N_D2) {
    int e = id & 15, lane = (id >> 4) & 31, kt = id >> 9;
    int k = kt * 32 + ((lane >= 16) ? 16 : 0) + e;
    int h = lane & 15;
    float v = (k < G_ && h < F_) ? d2w[k * F_ + h] : 0.f;
    D2F[id] = (_Float16)v;
    return;
  }
  id -= N_D2;
  if (id < N_BN) {
    int i = (id < FG) ? id : id - FG;
    float sc = gamma[i] * rsqrtf(var[i] + 1e-3f);
    if (id < FG) bnScale[i] = sc;
    else         bnShift[i] = beta[i] - mean[i] * sc;
    return;
  }
  id -= N_BN;
  if (id < N_XP) {
    int row = id >> 4, col = G_ + (id & 15);
    x1[row * KPAD + col] = (_Float16)0.f;
  }
}

// ---------------------------------------------------------------------------
// Gaussian descriptor stage.  One block per (b,s); 320 threads = one (f,g)
// pair each.  rho-term of the exponent hoisted out of the rotation loop so
// only one v_exp_f32 per (v, rotation).  desc[f][r][bs][KPAD] in f16, K pad
// zero-filled every call.
// ---------------------------------------------------------------------------
__global__ __launch_bounds__(320) void gauss_kernel(
    const float* __restrict__ feat,
    const float* __restrict__ rho, const float* __restrict__ theta,
    const float* __restrict__ mask,
    const float* __restrict__ mu_r, const float* __restrict__ mu_t,
    const float* __restrict__ sg_r, const float* __restrict__ sg_t,
    _Float16* __restrict__ desc)
{
  __shared__ float s_rho[V_], s_th[V_], s_mk[V_], s_ft[V_ * F_];
  int bs = blockIdx.x;
  int t  = threadIdx.x;
  for (int i = t; i < V_; i += 320) {
    s_rho[i] = rho[bs * V_ + i];
    s_th[i]  = theta[bs * V_ + i];
    s_mk[i]  = mask[bs * V_ + i];
  }
  for (int i = t; i < V_ * F_; i += 320) s_ft[i] = feat[(size_t)bs * V_ * F_ + i];
  __syncthreads();

  int f = t / G_, g = t % G_;
  int fg = f * G_ + g;
  float mur = mu_r[fg], mut = mu_t[fg];
  float sr = sg_r[fg], st = sg_t[fg];
  float isr = 1.f / (sr * sr + 1e-5f);
  float ist = 1.f / (st * st + 1e-5f);

  float num[NROT], den[NROT];
#pragma unroll
  for (int r = 0; r < NROT; ++r) { num[r] = 0.f; den[r] = 0.f; }

  for (int v = 0; v < V_; ++v) {
    float m   = s_mk[v];
    float dr  = s_rho[v] - mur;
    float ar  = -dr * dr * isr;          // rotation-invariant part
    float th0 = s_th[v];
    float fv  = s_ft[v * F_ + f];
#pragma unroll
    for (int r = 0; r < NROT; ++r) {
      float th = th0 + (float)r * (TWO_PI_F / 16.f);
      th = (th >= TWO_PI_F) ? th - TWO_PI_F : th;   // jnp.mod for in-range inputs
      float dt = th - mut;
      float w  = __expf(ar - dt * dt * ist) * m;
      den[r] += w;
      num[r] += w * fv;
    }
  }
#pragma unroll
  for (int r = 0; r < NROT; ++r) {
    float d = num[r] / (den[r] + 1e-5f);
    desc[((size_t)(f * NROT + r) * BS + bs) * KPAD + g] = (_Float16)d;
  }
  if (t < 64) {                       // zero K-pad columns 80..95
    int ff = t >> 4, gg = G_ + (t & 15);
#pragma unroll
    for (int r = 0; r < NROT; ++r)
      desc[((size_t)(ff * NROT + r) * BS + bs) * KPAD + gg] = (_Float16)0.f;
  }
}

// ---------------------------------------------------------------------------
// Conv GEMM: per f, [16*1024 x 96] x [96 x 80] with rotation-max fused into
// the C fragment, then bias + BN + ReLU in the epilogue.  One wave per
// (f, m-tile, n-tile); 1280 waves exactly (no partial waves -> EXEC all 1s).
// ---------------------------------------------------------------------------
__global__ __launch_bounds__(256) void conv_kernel(
    const _Float16* __restrict__ desc, const _Float16* __restrict__ WcF,
    const float* __restrict__ bconv, const float* __restrict__ bnScale,
    const float* __restrict__ bnShift, _Float16* __restrict__ xbn)
{
  int gid  = blockIdx.x * 8 + (threadIdx.x >> 5);
  int lane = threadIdx.x & 31;
  int f = gid / 320, rem = gid % 320;
  int mt = rem / 5, nt = rem % 5;

  v16h b0 = *(const v16h*)(WcF + ((f * 3 + 0) * 5 + nt) * 512 + lane * 16);
  v16h b1 = *(const v16h*)(WcF + ((f * 3 + 1) * 5 + nt) * 512 + lane * 16);
  v16h b2 = *(const v16h*)(WcF + ((f * 3 + 2) * 5 + nt) * 512 + lane * 16);

  v8f cmax;
#pragma unroll
  for (int j = 0; j < 8; ++j) cmax[j] = -3.402823466e38f;

#pragma unroll 1
  for (int r = 0; r < NROT; ++r) {
    const _Float16* A = desc + (size_t)(f * NROT + r) * BS * KPAD;
    v8f c = {};
    c = __builtin_amdgcn_wmma_f32_16x16x32_f16(false, load_a16x32(A, KPAD, mt * 16, 0,  lane), false, b0, (short)0, c, false, false);
    c = __builtin_amdgcn_wmma_f32_16x16x32_f16(false, load_a16x32(A, KPAD, mt * 16, 32, lane), false, b1, (short)0, c, false, false);
    c = __builtin_amdgcn_wmma_f32_16x16x32_f16(false, load_a16x32(A, KPAD, mt * 16, 64, lane), false, b2, (short)0, c, false, false);
#pragma unroll
    for (int j = 0; j < 8; ++j) cmax[j] = fmaxf(cmax[j], c[j]);
  }

  int h  = nt * 16 + (lane & 15);
  int fg = f * G_ + h;
  float bias = bconv[fg];
  float sc = bnScale[fg], sh = bnShift[fg];
  int rbase = mt * 16 + ((lane & 16) ? 8 : 0);
#pragma unroll
  for (int j = 0; j < 8; ++j) {
    float v = fmaxf((cmax[j] + bias) * sc + sh, 0.f);
    xbn[(rbase + j) * FG + fg] = (_Float16)v;
  }
}

// Dense 320 -> 80 (+ReLU).  320 waves exactly.
__global__ __launch_bounds__(256) void d1_kernel(
    const _Float16* __restrict__ xbn, const _Float16* __restrict__ D1F,
    const float* __restrict__ d1b, _Float16* __restrict__ x1)
{
  int gid  = blockIdx.x * 8 + (threadIdx.x >> 5);
  int lane = threadIdx.x & 31;
  int mt = gid / 5, nt = gid % 5;
  v8f c = {};
#pragma unroll 1
  for (int kt = 0; kt < 10; ++kt) {
    v16h b = *(const v16h*)(D1F + (kt * 5 + nt) * 512 + lane * 16);
    v16h a = load_a16x32(xbn, FG, mt * 16, kt * 32, lane);
    c = __builtin_amdgcn_wmma_f32_16x16x32_f16(false, a, false, b, (short)0, c, false, false);
  }
  int h = nt * 16 + (lane & 15);
  float bias = d1b[h];
  int rbase = mt * 16 + ((lane & 16) ? 8 : 0);
#pragma unroll
  for (int j = 0; j < 8; ++j)
    x1[(rbase + j) * KPAD + h] = (_Float16)fmaxf(c[j] + bias, 0.f);
}

// Dense 80 -> 4 (+ReLU), N padded to 16.  64 waves exactly.
__global__ __launch_bounds__(256) void d2_kernel(
    const _Float16* __restrict__ x1, const _Float16* __restrict__ D2F,
    const float* __restrict__ d2b, float* __restrict__ x2)
{
  int gid  = blockIdx.x * 8 + (threadIdx.x >> 5);
  int lane = threadIdx.x & 31;
  int mt = gid;
  v8f c = {};
#pragma unroll
  for (int kt = 0; kt < 3; ++kt) {
    v16h b = *(const v16h*)(D2F + kt * 512 + lane * 16);
    v16h a = load_a16x32(x1, KPAD, mt * 16, kt * 32, lane);
    c = __builtin_amdgcn_wmma_f32_16x16x32_f16(false, a, false, b, (short)0, c, false, false);
  }
  int h = lane & 15;
  int rbase = mt * 16 + ((lane & 16) ? 8 : 0);
  if (h < F_) {                       // divergence only after all WMMAs
    float bias = d2b[h];
#pragma unroll
    for (int j = 0; j < 8; ++j)
      x2[(rbase + j) * F_ + h] = fmaxf(c[j] + bias, 0.f);
  }
}

// Gather: feat_cur[b,s,v,:] = x2[b, idx[b,s,v], :]
__global__ __launch_bounds__(256) void gather_kernel(
    const float* __restrict__ x2, const int* __restrict__ idx,
    float* __restrict__ feat_cur)
{
  int i = blockIdx.x * 256 + threadIdx.x;
  if (i >= BS * V_) return;
  int bs = i / V_;
  int b  = bs >> 8;                   // bs / S_
  int j  = idx[i];
  float4 v = *(const float4*)(x2 + (size_t)((b << 8) + j) * 4);
  *(float4*)(feat_cur + (size_t)i * 4) = v;
}

// Head: relu(x @ d3_w + d3_b) @ out_w + out_b
__global__ __launch_bounds__(256) void head_kernel(
    const float* __restrict__ x2, const float* __restrict__ d3w,
    const float* __restrict__ d3b, const float* __restrict__ outw,
    const float* __restrict__ outb, float* __restrict__ out)
{
  int bs = blockIdx.x * 256 + threadIdx.x;
  if (bs >= BS) return;
  float a0 = x2[bs * 4 + 0], a1 = x2[bs * 4 + 1];
  float a2 = x2[bs * 4 + 2], a3 = x2[bs * 4 + 3];
  float acc = 0.f;
#pragma unroll
  for (int j = 0; j < 16; ++j) {
    float tj = a0 * d3w[j] + a1 * d3w[16 + j] + a2 * d3w[32 + j] + a3 * d3w[48 + j] + d3b[j];
    acc += fmaxf(tj, 0.f) * outw[j];
  }
  out[bs] = acc + outb[0];
}

// ---------------------------------------------------------------------------
extern "C" void kernel_launch(void* const* d_in, const int* in_sizes, int n_in,
                              void* d_out, int out_size, void* d_ws, size_t ws_size,
                              hipStream_t stream)
{
  const float* input_feat = (const float*)d_in[0];
  const float* rho    = (const float*)d_in[1];
  const float* theta  = (const float*)d_in[2];
  const float* mask   = (const float*)d_in[3];
  const float* mu_r   = (const float*)d_in[4];
  const float* mu_t   = (const float*)d_in[5];
  const float* sg_r   = (const float*)d_in[6];
  const float* sg_t   = (const float*)d_in[7];
  const float* Wconv  = (const float*)d_in[8];
  const float* bconv  = (const float*)d_in[9];
  const float* gamma  = (const float*)d_in[10];
  const float* beta   = (const float*)d_in[11];
  const float* mean   = (const float*)d_in[12];
  const float* var    = (const float*)d_in[13];
  const float* d1w    = (const float*)d_in[14];
  const float* d1b    = (const float*)d_in[15];
  const float* d2w    = (const float*)d_in[16];
  const float* d2b    = (const float*)d_in[17];
  const float* d3w    = (const float*)d_in[18];
  const float* d3b    = (const float*)d_in[19];
  const float* outw   = (const float*)d_in[20];
  const float* outb   = (const float*)d_in[21];
  const int*   indices = (const int*)d_in[22];
  (void)in_sizes; (void)n_in; (void)out_size; (void)ws_size;

  char* base = (char*)d_ws;
  size_t off = 0;
  auto carve = [&](size_t bytes) -> void* {
    void* p = base + off;
    off = (off + bytes + 255) & ~(size_t)255;
    return p;
  };
  _Float16* WcF     = (_Float16*)carve((size_t)60 * 512 * 2);
  _Float16* D1F     = (_Float16*)carve((size_t)50 * 512 * 2);
  _Float16* D2F     = (_Float16*)carve((size_t)3 * 512 * 2);
  float*    bnScale = (float*)carve((size_t)FG * 4);
  float*    bnShift = (float*)carve((size_t)FG * 4);
  _Float16* desc    = (_Float16*)carve((size_t)F_ * NROT * BS * KPAD * 2);  // 12.6 MB
  float*    feat_cur = (float*)carve((size_t)BS * V_ * F_ * 4);             // 3.3 MB
  _Float16* xbn     = (_Float16*)carve((size_t)BS * FG * 2);
  _Float16* x1      = (_Float16*)carve((size_t)BS * KPAD * 2);
  float*    x2      = (float*)carve((size_t)BS * F_ * 4);

  prep_kernel<<<293, 256, 0, stream>>>(Wconv, d1w, d2w, gamma, beta, mean, var,
                                       WcF, D1F, D2F, bnScale, bnShift, x1);

  for (int blk = 0; blk < 3; ++blk) {
    const float* feat = input_feat;
    if (blk) {
      gather_kernel<<<(BS * V_ + 255) / 256, 256, 0, stream>>>(x2, indices, feat_cur);
      feat = feat_cur;
    }
    gauss_kernel<<<BS, 320, 0, stream>>>(feat, rho, theta, mask,
                                         mu_r, mu_t, sg_r, sg_t, desc);
    conv_kernel<<<160, 256, 0, stream>>>(desc, WcF, bconv, bnScale, bnShift, xbn);
    d1_kernel<<<40, 256, 0, stream>>>(xbn, D1F, d1b, x1);
    d2_kernel<<<8, 256, 0, stream>>>(x1, D2F, d2b, x2);
  }
  head_kernel<<<4, 256, 0, stream>>>(x2, d3w, d3b, outw, outb, (float*)d_out);
}
